// MLPConv_86474871538667
// MI455X (gfx1250) — matrix operand order repaired and checked by reading, not compiled, not run on hardware
//
#include <hip/hip_runtime.h>

typedef __attribute__((ext_vector_type(16))) _Float16 v16h;
typedef __attribute__((ext_vector_type(8)))  float    v8f;

#define HID 32
#define N2_LOG 16           // 256*256 grid points
#define N2 (1 << N2_LOG)
#define TILES_PER_WAVE 4    // 16-row tiles per wave
#define WAVES_PER_BLOCK 8   // 256 threads
#define ROWS_PER_BLOCK (WAVES_PER_BLOCK * TILES_PER_WAVE * 16)

__device__ __forceinline__ float elu_f(float x) {
    return x > 0.0f ? x : (__expf(x) - 1.0f);
}

__global__ __launch_bounds__(256) void mlpconv_wmma_kernel(
    const float* __restrict__ x,
    const float* __restrict__ W0, const float* __restrict__ b0,
    const float* __restrict__ W1, const float* __restrict__ b1,
    const float* __restrict__ W2, const float* __restrict__ b2,
    float* __restrict__ out, int total_rows)
{
    const int lane = threadIdx.x & 31;
    const int wave = threadIdx.x >> 5;
    const int h    = lane >> 4;     // lane half (0/1) per WMMA fragment layouts
    const int nl   = lane & 15;     // M index (A/C frags) or N sub-index (B/C frags)

    // ---- hoist ALL per-lane weights into registers (global reads are L2
    //      broadcasts, done once; no LDS => no ds/wait traffic in the loop) ----
    // A-fragment K map (16-bit A 16x32): element e -> K = (e<8) ? 8h+e : 8+8h+e
    float w0r[5][16];
    float b0r[16];
#pragma unroll
    for (int e = 0; e < 16; ++e) {
        const int k = (e < 8) ? (8 * h + e) : (8 + 8 * h + e);
        b0r[e] = b0[k];
#pragma unroll
        for (int c = 0; c < 5; ++c) w0r[c][e] = W0[c * HID + k];
    }

    // B fragments for W1 (two 16-col tiles).
    // B 32x16 f16 layout: same K map as A fragment:
    // element e maps to K = (e<8) ? 8h+e : 8+8h+e.
    v16h bfrag0, bfrag1;
#pragma unroll
    for (int e = 0; e < 16; ++e) {
        const int k = (e < 8) ? (8 * h + e) : (8 + 8 * h + e);
        bfrag0[e] = (_Float16)W1[k * HID + nl];
        bfrag1[e] = (_Float16)W1[k * HID + (16 + nl)];
    }
    const float bias1_0 = b1[nl];
    const float bias1_1 = b1[16 + nl];
    const float w2_0 = W2[nl];
    const float w2_1 = W2[16 + nl];
    const float bias2 = b2[0];

    const int tileBase = (blockIdx.x * WAVES_PER_BLOCK + wave) * TILES_PER_WAVE;

#pragma unroll
    for (int it = 0; it < TILES_PER_WAVE; ++it) {
        const int rowBase = (tileBase + it) << 4;      // 16 rows per tile
        if (rowBase >= total_rows) break;              // wave-uniform guard

        // ---- stencil gather: this lane's row (duplicated across lane halves) ----
        const int row = rowBase + nl;
        const int t = row >> N2_LOG;
        const int p = row & (N2 - 1);
        const int i2 = p >> 8, j2 = p & 255;
        const float* xt = x + ((size_t)t << N2_LOG);
        const int pl = (p - 1)   & (N2 - 1);            // numpy flat -1 quirk
        const int pu = (p - 256) & (N2 - 1);            // numpy flat -n quirk
        const int pr = (i2 << 8) | ((j2 + 1) & 255);    // proper column wrap
        const int pd = (p + 256) & (N2 - 1);            // row wrap == flat wrap
        const float px0 = xt[pl], px1 = xt[pu], px2 = xt[p],
                    px3 = xt[pr], px4 = xt[pd];

        // ---- layer 0 (K=5, FP32 VALU) written directly in A-fragment order ----
        v16h afrag;
#pragma unroll
        for (int e = 0; e < 16; ++e) {
            float acc = b0r[e];
            acc = fmaf(px0, w0r[0][e], acc);
            acc = fmaf(px1, w0r[1][e], acc);
            acc = fmaf(px2, w0r[2][e], acc);
            acc = fmaf(px3, w0r[3][e], acc);
            acc = fmaf(px4, w0r[4][e], acc);
            afrag[e] = (_Float16)elu_f(acc);
        }

        // ---- layer 1: two v_wmma_f32_16x16x32_f16, bias folded into C ----
        v8f c0, c1;
#pragma unroll
        for (int r = 0; r < 8; ++r) { c0[r] = bias1_0; c1[r] = bias1_1; }
        v8f d0 = __builtin_amdgcn_wmma_f32_16x16x32_f16(
            false, afrag, false, bfrag0, (short)0, c0, false, false);
        v8f d1 = __builtin_amdgcn_wmma_f32_16x16x32_f16(
            false, afrag, false, bfrag1, (short)0, c1, false, false);

        // ---- ELU + layer 2 partial dot (D layout: VGPR r -> row 8h+r, col nl) ----
        float part[8];
#pragma unroll
        for (int r = 0; r < 8; ++r) {
            part[r] = fmaf(elu_f(d0[r]), w2_0, elu_f(d1[r]) * w2_1);
        }
        // butterfly reduction across the 16 lanes of each half (xor masks < 16)
#pragma unroll
        for (int m = 1; m < 16; m <<= 1) {
#pragma unroll
            for (int r = 0; r < 8; ++r) {
                part[r] += __shfl_xor(part[r], m, 32);
            }
        }

        // lanes nl<8 of each half write rows 8h+0..8h+7 (16 outputs per tile)
        if (nl < 8) {
            float v = part[0];
#pragma unroll
            for (int r = 1; r < 8; ++r) v = (nl == r) ? part[r] : v;
            out[rowBase + 8 * h + nl] = v + bias2;
        }
    }
}

extern "C" void kernel_launch(void* const* d_in, const int* in_sizes, int n_in,
                              void* d_out, int out_size, void* d_ws, size_t ws_size,
                              hipStream_t stream) {
    const float* x  = (const float*)d_in[0];
    const float* W0 = (const float*)d_in[1];
    const float* b0 = (const float*)d_in[2];
    const float* W1 = (const float*)d_in[3];
    const float* b1 = (const float*)d_in[4];
    const float* W2 = (const float*)d_in[5];
    const float* b2 = (const float*)d_in[6];
    float* out = (float*)d_out;

    const int total_rows = in_sizes[0];   // T * 65536 flattened rows
    const int blocks = (total_rows + ROWS_PER_BLOCK - 1) / ROWS_PER_BLOCK;
    mlpconv_wmma_kernel<<<blocks, 256, 0, stream>>>(x, W0, b0, W1, b1, W2, b2,
                                                    out, total_rows);
}
